// RNN_0_2637109920533
// MI455X (gfx1250) — matrix operand order repaired and checked by reading, not compiled, not run on hardware
//
#include <hip/hip_runtime.h>

// Problem constants (from the reference)
#define TT 48
#define BB 4096
#define HH 256
#define G4 1024   // 4*H

typedef __attribute__((ext_vector_type(16))) _Float16 v16h;
typedef __attribute__((ext_vector_type(8)))  _Float16 v8h;
typedef __attribute__((ext_vector_type(8)))  float    v8f;

__device__ __forceinline__ float sigmoidf_(float x) {
    return 1.0f / (1.0f + __expf(-x));
}

// ---------------------------------------------------------------------------
// v1 = W_ih @ inp_W ;  v2 = W_ih @ inp_b + b_ih + b_hh   (rank-1 collapse of
// the whole input pipeline: gx[t,b,g] = u[t,b]*v1[g] + v2[g])
// ---------------------------------------------------------------------------
__global__ __launch_bounds__(256) void pre_vecs_kernel(
    const float* __restrict__ W_ih, const float* __restrict__ inp_W,
    const float* __restrict__ inp_b, const float* __restrict__ b_ih,
    const float* __restrict__ b_hh, float* __restrict__ v1,
    float* __restrict__ v2)
{
    int g = blockIdx.x * blockDim.x + threadIdx.x;
    if (g >= G4) return;
    const float* w = W_ih + g * HH;
    float s1 = 0.f, s2 = 0.f;
    for (int k = 0; k < HH; ++k) {
        float wv = w[k];
        s1 += wv * inp_W[k];
        s2 += wv * inp_b[k];
    }
    v1[g] = s1;
    v2[g] = s2 + b_ih[g] + b_hh[g];
}

// ---------------------------------------------------------------------------
// Wt[k][g] = (f16) W_hh[g][k]  — transposed fp16 copy so B-operand fragments
// are coalesced across lanes (B layout: lane%16<->N col, elements<->K rows).
// ---------------------------------------------------------------------------
__global__ __launch_bounds__(256) void pre_wt_kernel(
    const float* __restrict__ W_hh, _Float16* __restrict__ Wt)
{
    int o = blockIdx.x * blockDim.x + threadIdx.x;   // 0 .. HH*G4-1
    if (o >= HH * G4) return;
    int g = o & (G4 - 1);
    int k = o >> 10;
    Wt[o] = (_Float16)W_hh[g * HH + k];
}

// Zero h (fp16) and c (fp32) — workspace is poisoned, must init each call.
__global__ __launch_bounds__(256) void init_state_kernel(
    _Float16* __restrict__ h16, float* __restrict__ c)
{
    const int n = BB * HH;
    for (int i = blockIdx.x * blockDim.x + threadIdx.x; i < n;
         i += gridDim.x * blockDim.x) {
        h16[i] = (_Float16)0.0f;
        c[i]   = 0.0f;
    }
}

// ---------------------------------------------------------------------------
// One LSTM step: gates = u⊗v1 + v2 + h @ W_hhᵀ ; fused gate math + c/h update.
// Each wave: 32 batch rows x 16 hidden cols x 4 gates (8 x v8f accumulators);
// each B fragment (shared W_hh^T tile) is reused for two WMMAs.
// Block = 4 waves (128 batch rows); grid = (B/128, H/16) = (32, 16).
// ---------------------------------------------------------------------------
__global__ __launch_bounds__(128) void lstm_step_kernel(
    const _Float16* __restrict__ hin, _Float16* __restrict__ hout,
    float* __restrict__ c, float* __restrict__ h32,
    const _Float16* __restrict__ Wt,          // [HH][G4] fp16 (W_hh transposed)
    const float* __restrict__ v1, const float* __restrict__ v2,
    const float* __restrict__ u,              // inputs + t*B  [B]
    int last)
{
    const int wave    = threadIdx.x >> 5;
    const int lane    = threadIdx.x & 31;
    const int halfsel = lane >> 4;            // 0: lanes 0-15, 1: lanes 16-31
    const int mrow    = lane & 15;
    const int bb      = blockIdx.x * 128 + wave * 32;  // batch tile base (32 rows)
    const int jb      = blockIdx.y * 16;               // hidden-col tile base

    v8f acc[2][4];
#pragma unroll
    for (int m = 0; m < 2; ++m)
#pragma unroll
        for (int gi = 0; gi < 4; ++gi)
#pragma unroll
            for (int i = 0; i < 8; ++i) acc[m][gi][i] = 0.f;

    // A fragment source: row-major fp16 h. ISA 16-bit A 16x32 layout:
    // lane m (0-15): VGPR0-3 = K 0..7, VGPR4-7 = K 16..23 ; lanes 16-31: +8.
    const _Float16* hrow0 = hin + (size_t)(bb + mrow) * HH;
    const _Float16* hrow1 = hrow0 + (size_t)16 * HH;

    auto loadA = [&](const _Float16* p) {
        v16h a;
        v8h x0 = *(const v8h*)(p + halfsel * 8);
        v8h x1 = *(const v8h*)(p + 16 + halfsel * 8);
#pragma unroll
        for (int i = 0; i < 8; ++i) { a[i] = x0[i]; a[8 + i] = x1[i]; }
        return a;
    };
    auto loadB = [&](const _Float16* p) {
        v16h b;
#pragma unroll
        for (int i = 0; i < 8; ++i) {
            b[i]     = p[(size_t)(halfsel * 8 + i) * G4];
            b[8 + i] = p[(size_t)(16 + halfsel * 8 + i) * G4];
        }
        return b;
    };

#pragma unroll 2
    for (int kb = 0; kb < HH; kb += 32) {
        v16h a0 = loadA(hrow0 + kb);
        v16h a1 = loadA(hrow1 + kb);
        // B fragment: lane%16 <-> N col; elements <-> K rows (same K order as A).
        const _Float16* brow = Wt + (size_t)kb * G4 + jb + mrow;
#pragma unroll
        for (int gi = 0; gi < 4; ++gi) {
            v16h b = loadB(brow + gi * HH);
            acc[0][gi] = __builtin_amdgcn_wmma_f32_16x16x32_f16(false, a0, false, b,
                                                                (short)0, acc[0][gi], false, false);
            acc[1][gi] = __builtin_amdgcn_wmma_f32_16x16x32_f16(false, a1, false, b,
                                                                (short)0, acc[1][gi], false, false);
        }
    }

    // C/D layout: VGPR r -> M = r + 8*halfsel ; N = lane%16 (fixed per lane).
    const int ncol = jb + mrow;
    float cv1[4], cv2[4];
#pragma unroll
    for (int gi = 0; gi < 4; ++gi) {
        cv1[gi] = v1[gi * HH + ncol];
        cv2[gi] = v2[gi * HH + ncol];
    }

#pragma unroll
    for (int m = 0; m < 2; ++m) {
#pragma unroll
        for (int r = 0; r < 8; ++r) {
            int b = bb + m * 16 + halfsel * 8 + r;
            float ub = u[b];
            float pi = acc[m][0][r] + ub * cv1[0] + cv2[0];
            float pf = acc[m][1][r] + ub * cv1[1] + cv2[1];
            float pg = acc[m][2][r] + ub * cv1[2] + cv2[2];
            float po = acc[m][3][r] + ub * cv1[3] + cv2[3];
            float ig = sigmoidf_(pi);
            float fg = sigmoidf_(pf);
            float gg = tanhf(pg);
            float og = sigmoidf_(po);
            size_t idx = (size_t)b * HH + ncol;
            float cn = fg * c[idx] + ig * gg;
            c[idx] = cn;
            float hn = og * tanhf(cn);
            hout[idx] = (_Float16)hn;
            if (last) h32[idx] = hn;
        }
    }
}

// out[b] = h32[b,:] . out_W + out_b   — one wave per row, shuffle reduce.
__global__ __launch_bounds__(128) void out_proj_kernel(
    const float* __restrict__ h32, const float* __restrict__ out_W,
    const float* __restrict__ out_b, float* __restrict__ out)
{
    int lane = threadIdx.x & 31;
    int row  = blockIdx.x * 4 + (threadIdx.x >> 5);
    float s = 0.f;
    for (int k = lane; k < HH; k += 32)
        s += h32[(size_t)row * HH + k] * out_W[k];
#pragma unroll
    for (int off = 16; off > 0; off >>= 1)
        s += __shfl_down(s, off, 32);
    if (lane == 0) out[row] = s + out_b[0];
}

extern "C" void kernel_launch(void* const* d_in, const int* in_sizes, int n_in,
                              void* d_out, int out_size, void* d_ws, size_t ws_size,
                              hipStream_t stream)
{
    (void)in_sizes; (void)n_in; (void)out_size; (void)ws_size;

    const float* inputs = (const float*)d_in[0];   // [T,B,1]
    const float* inp_W  = (const float*)d_in[1];   // [H,1]
    const float* inp_b  = (const float*)d_in[2];   // [H]
    const float* W_ih   = (const float*)d_in[3];   // [4H,H]
    const float* W_hh   = (const float*)d_in[4];   // [4H,H]
    const float* b_ih   = (const float*)d_in[5];   // [4H]
    const float* b_hh   = (const float*)d_in[6];   // [4H]
    const float* out_W  = (const float*)d_in[7];   // [1,H]
    const float* out_b  = (const float*)d_in[8];   // [1]
    float* out = (float*)d_out;                    // [B,1]

    // Workspace layout (~12.5 MB total)
    char* ws = (char*)d_ws;
    float*    v1  = (float*)ws;                    //  4 KB
    float*    v2  = v1 + G4;                       //  4 KB
    _Float16* Wt  = (_Float16*)(v2 + G4);          // 512 KB  (W_hh^T fp16)
    _Float16* hA  = Wt + HH * G4;                  //   2 MB
    _Float16* hB  = hA + BB * HH;                  //   2 MB
    float*    c   = (float*)(hB + BB * HH);        //   4 MB
    float*    h32 = c + BB * HH;                   //   4 MB

    pre_vecs_kernel<<<G4 / 256, 256, 0, stream>>>(W_ih, inp_W, inp_b, b_ih, b_hh, v1, v2);
    pre_wt_kernel<<<(HH * G4) / 256, 256, 0, stream>>>(W_hh, Wt);
    init_state_kernel<<<512, 256, 0, stream>>>(hA, c);

    dim3 grid(BB / 128, HH / 16);   // (32, 16)
    for (int t = 0; t < TT; ++t) {
        const _Float16* src = (t & 1) ? hB : hA;
        _Float16*       dst = (t & 1) ? hA : hB;
        lstm_step_kernel<<<grid, 128, 0, stream>>>(src, dst, c, h32, Wt, v1, v2,
                                                   inputs + (size_t)t * BB,
                                                   (t == TT - 1) ? 1 : 0);
    }

    out_proj_kernel<<<BB / 4, 128, 0, stream>>>(h32, out_W, out_b, out);
}
